// NSDE_87557203296836
// MI455X (gfx1250) — matrix operand order repaired and checked by reading, not compiled, not run on hardware
//
#include <hip/hip_runtime.h>

// ---------------------------------------------------------------------------
// Neural SDE Euler-Maruyama rollout on gfx1250 (MI455X).
//   * pack kernel: weights -> bf16, pre-swizzled into V_WMMA_F32_16X16X32_BF16
//     B-operand lane/VGPR layout in d_ws (L2 resident, ~1.25 MB).
//   * persistent kernel: 64 blocks x 512 threads (16 waves). Block g owns
//     batch rows [16g,16g+16); the 127-step recurrence runs entirely in-block
//     with LDS state. Register-blocked WMMA GEMMs, K-outer / tile-inner.
//   * sigma einsum reduced with __shfl_xor butterflies on the WMMA C fragment.
// ---------------------------------------------------------------------------

typedef __attribute__((ext_vector_type(16))) __bf16    v16bf;
typedef __attribute__((ext_vector_type(8)))  float     v8f;
typedef __attribute__((ext_vector_type(8)))  unsigned  v8u;

#define BB      1024
#define NNY     64
#define CCTL    16
#define MMW     16
#define HH      512
#define TT      128
#define KPAD    96
#define DTSTEP  0.01f
#define NTHREADS 512

// d_ws dword segment offsets (bf16 weight tiles, 256 dwords per 32x16 tile,
// laid out [tile][lane][vgpr]):
#define SEG_W1MU 0         // 3 ktiles * 32 ntiles = 96 tiles
#define SEG_W1SG 24576     // 96 tiles
#define SEG_W2MU 49152     // 16 * 4  = 64 tiles
#define SEG_W2SG 65536     // 16 * 64 = 1024 tiles
#define WS_DWORDS 327680   // 1.25 MB

__device__ __forceinline__ unsigned short f2bf(float f) {
  unsigned u = __builtin_bit_cast(unsigned, f);
  u += 0x7FFFu + ((u >> 16) & 1u);          // round-to-nearest-even
  return (unsigned short)(u >> 16);
}

__device__ __forceinline__ v8f wmma_bf16(v16bf a, v16bf b, v8f c) {
  return __builtin_amdgcn_wmma_f32_16x16x32_bf16(
      false, a, false, b, (short)0, c, false, false);
}

// A fragment (16x32 bf16) from row-major LDS matrix; S = row stride in ushorts.
__device__ __forceinline__ v16bf ldsA(const unsigned short* base, int S, int k0) {
  int lane = threadIdx.x & 31;
  const unsigned short* p = base + (lane & 15) * S + k0 + (lane >> 4) * 8;
  uint4 x0 = *(const uint4*)p;             // ds_load_b128
  uint4 x1 = *(const uint4*)(p + 16);      // ds_load_b128
  v8u u;
  u[0] = x0.x; u[1] = x0.y; u[2] = x0.z; u[3] = x0.w;
  u[4] = x1.x; u[5] = x1.y; u[6] = x1.z; u[7] = x1.w;
  return __builtin_bit_cast(v16bf, u);
}

// B fragment (32x16 bf16) from pre-swizzled global tiles.
__device__ __forceinline__ v16bf ldgB(const unsigned* __restrict__ seg, int tile) {
  int lane = threadIdx.x & 31;
  const unsigned* p = seg + (size_t)tile * 256 + lane * 8;
  uint4 x0 = *(const uint4*)p;             // global_load_b128
  uint4 x1 = *(const uint4*)(p + 4);       // global_load_b128
  v8u u;
  u[0] = x0.x; u[1] = x0.y; u[2] = x0.z; u[3] = x0.w;
  u[4] = x1.x; u[5] = x1.y; u[6] = x1.z; u[7] = x1.w;
  return __builtin_bit_cast(v16bf, u);
}

// ---------------------------------------------------------------------------
// Weight packer: f32 [K,N] -> bf16 tiles in WMMA B-operand layout (K padded).
// ---------------------------------------------------------------------------
__global__ void nsde_pack(const float* __restrict__ W1mu, const float* __restrict__ W1sg,
                          const float* __restrict__ W2mu, const float* __restrict__ W2sg,
                          unsigned* __restrict__ ws) {
  int d = blockIdx.x * blockDim.x + threadIdx.x;
  if (d >= WS_DWORDS) return;
  const float* W; int Kreal, Ncols, NT, base;
  if (d < SEG_W1SG)      { W = W1mu; Kreal = 80;  Ncols = 512;  NT = 32; base = SEG_W1MU; }
  else if (d < SEG_W2MU) { W = W1sg; Kreal = 80;  Ncols = 512;  NT = 32; base = SEG_W1SG; }
  else if (d < SEG_W2SG) { W = W2mu; Kreal = 512; Ncols = 64;   NT = 4;  base = SEG_W2MU; }
  else                   { W = W2sg; Kreal = 512; Ncols = 1024; NT = 64; base = SEG_W2SG; }
  int ld    = d - base;
  int t     = ld >> 8;
  int lane  = (ld >> 3) & 31;
  int v     = ld & 7;
  int ktile = t / NT, ntile = t % NT;
  int n  = ntile * 16 + (lane & 15);
  int k0 = ktile * 32 + ((lane & 16) ? 8 : 0) + v * 2 + ((v >= 4) ? 8 : 0);
  float f0 = (k0     < Kreal) ? W[(size_t)k0 * Ncols + n]       : 0.f;
  float f1 = (k0 + 1 < Kreal) ? W[(size_t)(k0 + 1) * Ncols + n] : 0.f;
  ws[d] = (unsigned)f2bf(f0) | ((unsigned)f2bf(f1) << 16);
}

// ---------------------------------------------------------------------------
// Persistent rollout kernel: one block = 16 batch rows for all 127 steps.
// ---------------------------------------------------------------------------
__global__ __launch_bounds__(NTHREADS) void nsde_rollout(
    const float* __restrict__ y0, const float* __restrict__ controls,
    const float* __restrict__ noise,
    const float* __restrict__ b1mu, const float* __restrict__ b2mu,
    const float* __restrict__ b1sg, const float* __restrict__ b2sg,
    const unsigned* __restrict__ ws, float* __restrict__ out) {

  __shared__ __align__(16) unsigned short x_lds[16 * KPAD];     // bf16 [16][96]
  __shared__ __align__(16) unsigned short h_lds[2 * 16 * HH];   // bf16 mu|sg [16][512]
  __shared__ __align__(16) float y_lds[16 * NNY];
  __shared__ __align__(16) float drift_lds[16 * NNY];
  __shared__ __align__(16) float contrib_lds[16 * NNY];
  __shared__ __align__(16) float dW_lds[16 * MMW];
  __shared__ __align__(16) float b1mu_l[HH], b1sg_l[HH];
  __shared__ __align__(16) float b2mu_l[NNY], b2sg_l[NNY * MMW];

  const int tid  = threadIdx.x;
  const int W    = tid >> 5;                 // wave 0..15
  const int lane = tid & 31;
  const int lan  = lane & 15;
  const int hi   = lane >> 4;
  const int b0   = blockIdx.x * 16;          // 64 blocks
  const float sqrt_dt = sqrtf(DTSTEP);

  const unsigned* segW1mu = ws + SEG_W1MU;
  const unsigned* segW1sg = ws + SEG_W1SG;
  const unsigned* segW2mu = ws + SEG_W2MU;
  const unsigned* segW2sg = ws + SEG_W2SG;

  // ---- init: y0 -> LDS + out[0]; biases -> LDS ----
  for (int i = tid; i < 16 * NNY; i += NTHREADS) {
    int bl = i >> 6, n = i & 63;
    float v = y0[(size_t)(b0 + bl) * NNY + n];
    y_lds[i] = v;
    out[(size_t)(b0 + bl) * NNY + n] = v;    // trajectory t = 0
  }
  for (int i = tid; i < HH; i += NTHREADS) { b1mu_l[i] = b1mu[i]; b1sg_l[i] = b1sg[i]; }
  if (tid < NNY) b2mu_l[tid] = b2mu[tid];
  for (int i = tid; i < NNY * MMW; i += NTHREADS) b2sg_l[i] = b2sg[i];
  __syncthreads();

  for (int t = 0; t < TT - 1; ++t) {
    // ---- phase 0: x = concat(y, u, 0-pad) in bf16 ; dW = z*sqrt(dt) ----
    for (int i = tid; i < 16 * KPAD; i += NTHREADS) {
      int r = i / KPAD, c = i % KPAD;
      float v = (c < NNY) ? y_lds[r * NNY + c]
              : (c < NNY + CCTL ? controls[(size_t)t * CCTL + (c - NNY)] : 0.f);
      x_lds[i] = f2bf(v);
    }
    if (tid < 256) {
      int r = tid >> 4, m = tid & 15;
      dW_lds[tid] = noise[((size_t)t * BB + (b0 + r)) * MMW + m] * sqrt_dt;
    }
    __syncthreads();

    // ---- phase 1: hidden = relu(x @ W1 + b1); 4 n-tiles per wave ----
    {
      const v16bf a0 = ldsA(x_lds, KPAD, 0);
      const v16bf a1 = ldsA(x_lds, KPAD, 32);
      const v16bf a2 = ldsA(x_lds, KPAD, 64);
      const int mlp = W >> 3;                       // waves 0-7 mu, 8-15 sg
      const unsigned* seg = mlp ? segW1sg : segW1mu;
      const float* bsrc   = mlp ? b1sg_l  : b1mu_l;
      unsigned short* h   = h_lds + mlp * 16 * HH;
      const int nt0 = (W & 7) * 4;
      v8f acc[4] = {v8f{}, v8f{}, v8f{}, v8f{}};
#pragma unroll
      for (int i = 0; i < 4; ++i) {
        acc[i] = wmma_bf16(a0, ldgB(seg,      nt0 + i), acc[i]);
        acc[i] = wmma_bf16(a1, ldgB(seg, 32 + nt0 + i), acc[i]);
        acc[i] = wmma_bf16(a2, ldgB(seg, 64 + nt0 + i), acc[i]);
      }
#pragma unroll
      for (int i = 0; i < 4; ++i) {
        int n = (nt0 + i) * 16 + lan;
        float bias = bsrc[n];
#pragma unroll
        for (int r = 0; r < 8; ++r)
          h[(r + 8 * hi) * HH + n] = f2bf(fmaxf(acc[i][r] + bias, 0.f));
      }
    }
    __syncthreads();

    // ---- phase 2a: sigma GEMM2, register-blocked: wave W owns sg n-tiles
    //      [4W, 4W+4); K outer so the A fragment is loaded once per kt ----
    {
      const unsigned short* hA = h_lds + 16 * HH;   // h_sg
      const int nt0 = W * 4;
      v8f acc[4] = {v8f{}, v8f{}, v8f{}, v8f{}};
      for (int kt = 0; kt < 16; ++kt) {
        v16bf a = ldsA(hA, HH, kt * 32);
        const int tb = kt * 64 + nt0;
#pragma unroll
        for (int i = 0; i < 4; ++i)
          acc[i] = wmma_bf16(a, ldgB(segW2sg, tb + i), acc[i]);
      }
      // einsum epilogue: sg tile n covers diff[:, n, m=0..15]; lane holds m.
#pragma unroll
      for (int i = 0; i < 4; ++i) {
        const int n = nt0 + i;
        float bias = b2sg_l[n * MMW + lan];
#pragma unroll
        for (int r = 0; r < 8; ++r) {
          int bl = r + 8 * hi;
          float v = (acc[i][r] + bias) * dW_lds[bl * MMW + lan];
          v += __shfl_xor(v, 1, 32);
          v += __shfl_xor(v, 2, 32);
          v += __shfl_xor(v, 4, 32);
          v += __shfl_xor(v, 8, 32);
          if (lan == 0) contrib_lds[bl * NNY + n] = v;
        }
      }
    }

    // ---- phase 2b: drift GEMM2 tail on waves 0-3 (wave-uniform branch) ----
    if (W < 4) {
      v8f acc = {};
      for (int kt = 0; kt < 16; ++kt) {
        v16bf a = ldsA(h_lds, HH, kt * 32);         // h_mu
        acc = wmma_bf16(a, ldgB(segW2mu, kt * 4 + W), acc);
      }
      int n = W * 16 + lan;
      float bias = b2mu_l[n];
#pragma unroll
      for (int r = 0; r < 8; ++r)
        drift_lds[(r + 8 * hi) * NNY + n] = acc[r] + bias;
    }
    __syncthreads();

    // ---- phase 3: y += drift*dt + einsum; emit trajectory row t+1 ----
    for (int i = tid; i < 16 * NNY; i += NTHREADS) {
      int bl = i >> 6, n = i & 63;
      float yn = y_lds[i] + drift_lds[i] * DTSTEP + contrib_lds[i];
      y_lds[i] = yn;
      out[((size_t)(t + 1) * BB + (b0 + bl)) * NNY + n] = yn;
    }
    __syncthreads();
  }
}

// ---------------------------------------------------------------------------
extern "C" void kernel_launch(void* const* d_in, const int* in_sizes, int n_in,
                              void* d_out, int out_size, void* d_ws, size_t ws_size,
                              hipStream_t stream) {
  const float* y0       = (const float*)d_in[0];
  const float* controls = (const float*)d_in[1];
  const float* noise    = (const float*)d_in[2];
  const float* W1_mu    = (const float*)d_in[3];
  const float* b1_mu    = (const float*)d_in[4];
  const float* W2_mu    = (const float*)d_in[5];
  const float* b2_mu    = (const float*)d_in[6];
  const float* W1_sg    = (const float*)d_in[7];
  const float* b1_sg    = (const float*)d_in[8];
  const float* W2_sg    = (const float*)d_in[9];
  const float* b2_sg    = (const float*)d_in[10];
  float* out            = (float*)d_out;
  unsigned* ws          = (unsigned*)d_ws;

  nsde_pack<<<(WS_DWORDS + 255) / 256, 256, 0, stream>>>(W1_mu, W1_sg, W2_mu, W2_sg, ws);

  nsde_rollout<<<BB / 16, NTHREADS, 0, stream>>>(y0, controls, noise,
                                                 b1_mu, b2_mu, b1_sg, b2_sg,
                                                 ws, out);
}
